// Pointnet2_21827023798635
// MI455X (gfx1250) — matrix operand order repaired and checked by reading, not compiled, not run on hardware
//
#include <hip/hip_runtime.h>
#include <hip/hip_bf16.h>

// ---------------------------------------------------------------------------
// PointNet++ forward for MI455X (gfx1250).  All dense linear layers run through
// a wave32 WMMA GEMM (v_wmma_f32_16x16x32_f16, fp32 accumulate).  Activations
// and weights are pre-packed to f16 outside the GEMM hot loop, so tile staging
// is pure b128 copies; fragments load as contiguous ds_load_b128 (k = 16h + s, same in A and B).
// BN uses data statistics (fused bias + sum/sumsq epilogue) like the reference.
// ---------------------------------------------------------------------------

#define BB 8
#define N0 8192

typedef __attribute__((ext_vector_type(16))) _Float16 v16h;
typedef __attribute__((ext_vector_type(8)))  _Float16 v8h;
typedef __attribute__((ext_vector_type(8)))  float    v8f;

// ---------------- utility kernels ----------------

__global__ void k_zero(float* p, int n) {
    int i = blockIdx.x * blockDim.x + threadIdx.x;
    if (i < n) p[i] = 0.f;
}

// xyz (B,3,N) -> pts0 (B,N,3)
__global__ void k_transpose_in(const float* __restrict__ in, float* __restrict__ out,
                               int Bc, int Nn) {
    long i = (long)blockIdx.x * blockDim.x + threadIdx.x;
    long tot = (long)Bc * Nn * 3;
    if (i >= tot) return;
    int c = (int)(i % 3);
    int n = (int)((i / 3) % Nn);
    int b = (int)(i / (3L * Nn));
    out[i] = in[((long)b * 3 + c) * Nn + n];
}

// l0 (B*N,128) -> (B,128,N)
__global__ void k_transpose_out(const float* __restrict__ l0, float* __restrict__ out,
                                int Bc, int Nn, int C) {
    long i = (long)blockIdx.x * blockDim.x + threadIdx.x;
    long tot = (long)Bc * C * Nn;
    if (i >= tot) return;
    int n = (int)(i % Nn);
    int c = (int)((i / Nn) % C);
    int b = (int)(i / ((long)C * Nn));
    out[i] = l0[((long)b * Nn + n) * C + c];
}

// ---------------- farthest point sampling ----------------
__global__ void k_fps(const float* __restrict__ xyz, int Nl, int npoint,
                      float* __restrict__ dist, int* __restrict__ fpsidx) {
    int b = blockIdx.x;
    const float* P = xyz + (size_t)b * Nl * 3;
    float* dl = dist + (size_t)b * Nl;
    __shared__ float cx, cy, cz;
    __shared__ float smax[256];
    __shared__ int   simax[256];
    __shared__ int   sfar;
    int t = threadIdx.x;
    for (int i = t; i < Nl; i += 256) dl[i] = 1e10f;
    if (t == 0) sfar = 0;
    __syncthreads();
    for (int it = 0; it < npoint; ++it) {
        if (t == 0) {
            int f = sfar;
            fpsidx[b * npoint + it] = f;
            cx = P[f * 3 + 0]; cy = P[f * 3 + 1]; cz = P[f * 3 + 2];
        }
        __syncthreads();
        float best = -1.f; int bi = 0;
        for (int i = t; i < Nl; i += 256) {
            float dx = P[i * 3 + 0] - cx;
            float dy = P[i * 3 + 1] - cy;
            float dz = P[i * 3 + 2] - cz;
            float d  = dx * dx + dy * dy + dz * dz;
            float dm = fminf(dl[i], d);
            dl[i] = dm;
            if (dm > best) { best = dm; bi = i; }
        }
        smax[t] = best; simax[t] = bi;
        __syncthreads();
        for (int s = 128; s > 0; s >>= 1) {
            if (t < s) {
                if (smax[t + s] > smax[t] ||
                    (smax[t + s] == smax[t] && simax[t + s] < simax[t])) {
                    smax[t] = smax[t + s]; simax[t] = simax[t + s];
                }
            }
            __syncthreads();
        }
        if (t == 0) sfar = simax[0];
        __syncthreads();
    }
}

__global__ void k_gather_newxyz(const float* __restrict__ xyz, const int* __restrict__ fpsidx,
                                int S, int Nl, float* __restrict__ new_xyz, int Bc) {
    int i = blockIdx.x * blockDim.x + threadIdx.x;
    if (i >= Bc * S) return;
    int b = i / S, s = i % S;
    int j = fpsidx[b * S + s];
    const float* p = xyz + ((size_t)b * Nl + j) * 3;
    new_xyz[(size_t)i * 3 + 0] = p[0];
    new_xyz[(size_t)i * 3 + 1] = p[1];
    new_xyz[(size_t)i * 3 + 2] = p[2];
}

// first-K indices within radius (matches top_k over masked index array)
__global__ void k_ballquery(const float* __restrict__ xyz, const float* __restrict__ new_xyz,
                            int Nl, int S, int K, float r2, int* __restrict__ gidx, int Bc) {
    int i = blockIdx.x * blockDim.x + threadIdx.x;
    if (i >= Bc * S) return;
    int b = i / S;
    const float* c = new_xyz + (size_t)i * 3;
    const float* P = xyz + (size_t)b * Nl * 3;
    int* g = gidx + (size_t)i * K;
    int cnt = 0, first = 0;
    for (int j = 0; j < Nl && cnt < K; ++j) {
        float dx = P[j * 3 + 0] - c[0];
        float dy = P[j * 3 + 1] - c[1];
        float dz = P[j * 3 + 2] - c[2];
        float d = dx * dx + dy * dy + dz * dz;
        if (d <= r2) {
            if (cnt == 0) first = j;
            g[cnt++] = j;
        }
    }
    for (; cnt < K; ++cnt) g[cnt] = first;
}

// grouped features (f16): [xyz[g]-new_xyz , pts[g] , zero-pad] -> A16 (B*S*K, Cpad)
__global__ void k_group(const float* __restrict__ xyz, const float* __restrict__ pts,
                        const float* __restrict__ new_xyz, const int* __restrict__ gidx,
                        int Nl, int S, int K, int Cl, int Cpad,
                        _Float16* __restrict__ A, int Bc) {
    long row = (long)blockIdx.x * blockDim.x + threadIdx.x;
    long tot = (long)Bc * S * K;
    if (row >= tot) return;
    int k = (int)(row % K);
    long t2 = row / K;
    int s = (int)(t2 % S);
    int b = (int)(t2 / S);
    int j = gidx[((long)b * S + s) * K + k];
    const float* p  = xyz + ((size_t)b * Nl + j) * 3;
    const float* nz = new_xyz + ((size_t)b * S + s) * 3;
    _Float16* o = A + (size_t)row * Cpad;
    o[0] = (_Float16)(p[0] - nz[0]);
    o[1] = (_Float16)(p[1] - nz[1]);
    o[2] = (_Float16)(p[2] - nz[2]);
    const float* f = pts + ((size_t)b * Nl + j) * Cl;
    for (int c = 0; c < Cl; ++c) o[3 + c] = (_Float16)f[c];
    for (int c = 3 + Cl; c < Cpad; ++c) o[c] = (_Float16)0.f;
}

// pack weights: W (Kact x N, f32) -> Wp (Npad x Kpad, f16, transposed, zero-pad)
__global__ void k_packw(const float* __restrict__ W, _Float16* __restrict__ Wp,
                        int Kact, int Kpad, int N, int Npad) {
    int i = blockIdx.x * blockDim.x + threadIdx.x;
    int tot = Npad * Kpad;
    if (i >= tot) return;
    int k = i % Kpad, n = i / Kpad;
    Wp[i] = (k < Kact && n < N) ? (_Float16)W[(size_t)k * N + n] : (_Float16)0.f;
}

// ---------------- WMMA GEMM + bias + per-channel stats ----------------
// block = 128 threads = 4 waves; block tile = 64 rows x 32 cols;
// each wave: one 16-row subtile x two 16-col subtiles (A fragment reused).
// A16 row-major (stride Kpad), Wp pre-transposed f16 (stride Kpad).
__global__ void __launch_bounds__(128) k_gemm_bias(
    const _Float16* __restrict__ A16, const _Float16* __restrict__ Wp,
    const float* __restrict__ bias, float* __restrict__ Y,
    int M, int N, int Kpad,
    float* __restrict__ sums, float* __restrict__ sumsq) {
    __shared__ _Float16 As[64][40];    // 64 rows x 32 k (row stride 80B, 16B-aligned)
    __shared__ _Float16 BsT[32][40];   // 32 n  x 32 k
    const int t    = threadIdx.x;
    const int lane = t & 31;
    const int wave = t >> 5;
    const int rowBase = blockIdx.x * 64;
    const int colBase = blockIdx.y * 32;
    const int mloc = lane & 15;
    const int half = lane >> 4;

    // per-thread staging coordinates (pure b128 copies, no guards, no cvt)
    const int ra0 = t >> 2,        ca0 = (t & 3) * 8;          // A chunk 0
    const int ra1 = (128 + t) >> 2, ca1 = ((128 + t) & 3) * 8; // A chunk 1
    const int nb  = t >> 2,        cb  = (t & 3) * 8;          // B chunk

    v8f acc0 = {}, acc1 = {};
    for (int kb = 0; kb < Kpad; kb += 32) {
        // issue all three b128 loads first so they can clause
        v8h a0 = *(const v8h*)&A16[(size_t)(rowBase + ra0) * Kpad + kb + ca0];
        v8h a1 = *(const v8h*)&A16[(size_t)(rowBase + ra1) * Kpad + kb + ca1];
        v8h bw = *(const v8h*)&Wp[(size_t)(colBase + nb) * Kpad + kb + cb];
        if (kb + 32 < Kpad)  // speculative prefetch of next A tile
            __builtin_prefetch(&A16[(size_t)(rowBase + ra0) * Kpad + kb + 32 + ca0], 0, 1);
        *(v8h*)&As[ra0][ca0] = a0;
        *(v8h*)&As[ra1][ca1] = a1;
        *(v8h*)&BsT[nb][cb]  = bw;
        __syncthreads();

        // A fragment: two contiguous 16B runs of one LDS row -> 2x ds_load_b128
        const _Float16* arow = &As[wave * 16 + mloc][0];
        v8h alo = *(const v8h*)(arow + half * 16);
        v8h ahi = *(const v8h*)(arow + half * 16 + 8);
        v16h af = __builtin_shufflevector(alo, ahi, 0,1,2,3,4,5,6,7,8,9,10,11,12,13,14,15);

        // B fragments: k in [half*16, half*16+16) of BsT row n -> 2x ds_load_b128
        const _Float16* brow0 = &BsT[mloc][0];
        v8h b0lo = *(const v8h*)(brow0 + half * 16);
        v8h b0hi = *(const v8h*)(brow0 + half * 16 + 8);
        v16h bf0 = __builtin_shufflevector(b0lo, b0hi, 0,1,2,3,4,5,6,7,8,9,10,11,12,13,14,15);
        const _Float16* brow1 = &BsT[16 + mloc][0];
        v8h b1lo = *(const v8h*)(brow1 + half * 16);
        v8h b1hi = *(const v8h*)(brow1 + half * 16 + 8);
        v16h bf1 = __builtin_shufflevector(b1lo, b1hi, 0,1,2,3,4,5,6,7,8,9,10,11,12,13,14,15);

        acc0 = __builtin_amdgcn_wmma_f32_16x16x32_f16(
            false, af, false, bf0, (short)0, acc0, false, false);
        acc1 = __builtin_amdgcn_wmma_f32_16x16x32_f16(
            false, af, false, bf1, (short)0, acc1, false, false);
        __syncthreads();
    }

    // epilogue: bias + store + per-channel partial sums (two columns per lane)
    #pragma unroll
    for (int nt = 0; nt < 2; ++nt) {
        int col = colBase + nt * 16 + mloc;
        if (col < N) {
            float bv = bias[col];
            float s1 = 0.f, s2 = 0.f;
            #pragma unroll
            for (int i = 0; i < 8; ++i) {
                int row = rowBase + wave * 16 + half * 8 + i;   // C/D layout
                float y = (nt ? acc1[i] : acc0[i]) + bv;
                Y[(size_t)row * N + col] = y;
                s1 += y; s2 += y * y;
            }
            atomicAdd(&sums[col], s1);
            atomicAdd(&sumsq[col], s2);
        }
    }
}

// stats -> scale/shift  (biased variance like jnp.var)
__global__ void k_bn_finalize(float* stats, const float* __restrict__ g,
                              const float* __restrict__ be, int M, int N) {
    int c = blockIdx.x * blockDim.x + threadIdx.x;
    if (c >= N) return;
    float mean = stats[c] / (float)M;
    float var  = stats[512 + c] / (float)M - mean * mean;
    float sc   = g[c] * rsqrtf(var + 1e-5f);
    stats[1024 + c] = sc;
    stats[1536 + c] = be[c] - mean * sc;
}

// BN scale/shift + ReLU; optional fp32 and f16 destinations
__global__ void k_bn_relu(const float* __restrict__ Y, const float* __restrict__ stats,
                          float* __restrict__ dstF, _Float16* __restrict__ dst16,
                          long total, int N) {
    long i = (long)blockIdx.x * blockDim.x + threadIdx.x;
    if (i >= total) return;
    int c = (int)(i % N);
    float v = fmaxf(0.f, Y[i] * stats[1024 + c] + stats[1536 + c]);
    if (dstF)  dstF[i]  = v;
    if (dst16) dst16[i] = (_Float16)v;
}

// max over the K (nsample) axis: (B*S, K, C) -> (B*S, C)
__global__ void k_maxpool(const float* __restrict__ X, int BS, int K, int C,
                          float* __restrict__ out) {
    long i = (long)blockIdx.x * blockDim.x + threadIdx.x;
    if (i >= (long)BS * C) return;
    int c  = (int)(i % C);
    long bs = i / C;
    float m = -3.4e38f;
    for (int k = 0; k < K; ++k)
        m = fmaxf(m, X[((size_t)bs * K + k) * C + c]);
    out[i] = m;
}

// 3-NN inverse-distance interpolation + optional concat; f16 output for GEMM
__global__ void k_fp_interp(const float* __restrict__ xyz1, const float* __restrict__ xyz2,
                            const float* __restrict__ pts1, const float* __restrict__ pts2,
                            int N1, int N2, int C1, int C2,
                            _Float16* __restrict__ out, int Bc) {
    int i = blockIdx.x * blockDim.x + threadIdx.x;
    if (i >= Bc * N1) return;
    int b = i / N1;
    const float* q = xyz1 + (size_t)i * 3;
    const float* P = xyz2 + (size_t)b * N2 * 3;
    float d0 = 3.4e38f, d1 = 3.4e38f, d2 = 3.4e38f;
    int i0 = 0, i1 = 0, i2 = 0;
    for (int j = 0; j < N2; ++j) {
        float dx = P[j * 3 + 0] - q[0];
        float dy = P[j * 3 + 1] - q[1];
        float dz = P[j * 3 + 2] - q[2];
        float d = dx * dx + dy * dy + dz * dz;
        if (d < d0)      { d2 = d1; i2 = i1; d1 = d0; i1 = i0; d0 = d; i0 = j; }
        else if (d < d1) { d2 = d1; i2 = i1; d1 = d;  i1 = j; }
        else if (d < d2) { d2 = d;  i2 = j; }
    }
    float r0 = 1.f / (d0 + 1e-8f), r1 = 1.f / (d1 + 1e-8f), r2 = 1.f / (d2 + 1e-8f);
    float sw = r0 + r1 + r2;
    r0 /= sw; r1 /= sw; r2 /= sw;
    _Float16* o = out + (size_t)i * (C1 + C2);
    if (pts1 != nullptr)
        for (int c = 0; c < C1; ++c) o[c] = (_Float16)pts1[(size_t)i * C1 + c];
    const float* p0 = pts2 + ((size_t)b * N2 + i0) * C2;
    const float* p1 = pts2 + ((size_t)b * N2 + i1) * C2;
    const float* p2 = pts2 + ((size_t)b * N2 + i2) * C2;
    for (int c = 0; c < C2; ++c)
        o[C1 + c] = (_Float16)(r0 * p0[c] + r1 * p1[c] + r2 * p2[c]);
}

// ---------------------------------------------------------------------------
extern "C" void kernel_launch(void* const* d_in, const int* in_sizes, int n_in,
                              void* d_out, int out_size, void* d_ws, size_t ws_size,
                              hipStream_t stream) {
    (void)in_sizes; (void)n_in; (void)out_size; (void)ws_size;
    const float* xyz_in = (const float*)d_in[0];

    // --- parameters in JAX tree-flatten (alphabetical) leaf order ---
    struct Layer { const float *W, *b, *be, *g; };
    int pi = 1;
    auto nextL = [&](Layer& L) {
        L.W  = (const float*)d_in[pi++];
        L.b  = (const float*)d_in[pi++];
        L.be = (const float*)d_in[pi++];
        L.g  = (const float*)d_in[pi++];
    };
    Layer fp1[3], fp2[2], fp3[2], fp4[2], head, sa[4][3];
    for (int i = 0; i < 3; ++i) nextL(fp1[i]);
    for (int i = 0; i < 2; ++i) nextL(fp2[i]);
    for (int i = 0; i < 2; ++i) nextL(fp3[i]);
    for (int i = 0; i < 2; ++i) nextL(fp4[i]);
    nextL(head);
    for (int l = 0; l < 4; ++l)
        for (int i = 0; i < 3; ++i) nextL(sa[l][i]);

    // --- workspace bump allocator ---
    char* wsb = (char*)d_ws;
    size_t off = 0;
    auto allocf = [&](size_t n) -> float* {
        float* p = (float*)(wsb + off);
        off = (off + n * sizeof(float) + 255) & ~(size_t)255;
        return p;
    };
    auto alloch = [&](size_t n) -> _Float16* {
        _Float16* p = (_Float16*)(wsb + off);
        off = (off + n * sizeof(_Float16) + 255) & ~(size_t)255;
        return p;
    };
    auto alloci = [&](size_t n) -> int* {
        int* p = (int*)(wsb + off);
        off = (off + n * sizeof(int) + 255) & ~(size_t)255;
        return p;
    };

    const int   Svals[4]    = {2048, 512, 128, 32};
    const float Rvals[4]    = {0.1f, 0.2f, 0.4f, 0.8f};
    const int   Cprev[4]    = {3, 64, 128, 256};
    const int   Cpad[4]     = {32, 96, 160, 288};
    const int   Chans[4][3] = {{32,32,64},{64,64,128},{128,128,256},{256,256,512}};

    float* pts0  = allocf((size_t)BB * N0 * 3);
    float* dist  = allocf((size_t)BB * N0);
    float* stats = allocf(2048);
    float* xs[5]; float* ps[5];
    xs[0] = pts0; ps[0] = pts0;
    for (int l = 0; l < 4; ++l) {
        xs[l + 1] = allocf((size_t)BB * Svals[l] * 3);
        ps[l + 1] = allocf((size_t)BB * Svals[l] * Chans[l][2]);
    }
    int* fpsidx = alloci((size_t)BB * 2048);
    int* gidx   = alloci((size_t)BB * 2048 * 32);
    _Float16* grouped16 = alloch((size_t)16777216);   // grouped / concat A16
    _Float16* h0   = alloch((size_t)33554432);        // f16 activation ping
    _Float16* h1   = alloch((size_t)33554432);        // f16 activation pong
    _Float16* wp16 = alloch((size_t)262144);          // packed weights
    _Float16* l0h  = alloch((size_t)BB * 8192 * 128); // l0 f16 (head input)
    float* yb   = allocf((size_t)33554432);           // GEMM output (pre-BN)
    float* actF = allocf((size_t)33554432);           // final fp32 activation (maxpool in)
    float* l3 = allocf((size_t)BB * 128 * 256);
    float* l2 = allocf((size_t)BB * 512 * 256);
    float* l1 = allocf((size_t)BB * 2048 * 128);
    float* l0 = allocf((size_t)BB * 8192 * 128);

    auto run_layer = [&](const _Float16* A16, int M, int Kact, int Kp, const Layer& L,
                         int Nout, float* dstF, _Float16* dst16) {
        int Npad = (Nout + 31) & ~31;
        k_packw<<<(Npad * Kp + 255) / 256, 256, 0, stream>>>(L.W, wp16, Kact, Kp, Nout, Npad);
        k_zero<<<4, 256, 0, stream>>>(stats, 1024);
        dim3 g(M / 64, Npad / 32);
        k_gemm_bias<<<g, 128, 0, stream>>>(A16, wp16, L.b, yb, M, Nout, Kp,
                                           stats, stats + 512);
        k_bn_finalize<<<2, 256, 0, stream>>>(stats, L.g, L.be, M, Nout);
        long total = (long)M * Nout;
        k_bn_relu<<<(int)((total + 255) / 256), 256, 0, stream>>>(yb, stats, dstF, dst16,
                                                                  total, Nout);
    };
    auto run_mlp = [&](const _Float16* A0, int M, int Kact0, int Kpad0,
                       const Layer* Ls, const int* chans, int nlayers,
                       float* finalF, _Float16* final16) {
        const _Float16* Acur = A0;
        int Kact = Kact0, Kp = Kpad0;
        for (int j = 0; j < nlayers; ++j) {
            bool last = (j == nlayers - 1);
            _Float16* d16 = last ? final16 : ((j & 1) ? h1 : h0);
            float* dF = last ? finalF : nullptr;
            run_layer(Acur, M, Kact, Kp, Ls[j], chans[j], dF, d16);
            Acur = d16;
            Kact = Kp = chans[j];
        }
    };

    // --- input transpose (B,3,N) -> (B,N,3) ---
    {
        long tot = (long)BB * N0 * 3;
        k_transpose_in<<<(int)((tot + 255) / 256), 256, 0, stream>>>(xyz_in, pts0, BB, N0);
    }

    // --- set-abstraction stages ---
    int Nl = N0;
    for (int l = 0; l < 4; ++l) {
        int S = Svals[l], K = 32, Cl = Cprev[l], Cp = Cpad[l];
        k_fps<<<BB, 256, 0, stream>>>(xs[l], Nl, S, dist, fpsidx);
        int bs = BB * S;
        k_gather_newxyz<<<(bs + 255) / 256, 256, 0, stream>>>(xs[l], fpsidx, S, Nl,
                                                              xs[l + 1], BB);
        k_ballquery<<<(bs + 255) / 256, 256, 0, stream>>>(xs[l], xs[l + 1], Nl, S, K,
                                                          Rvals[l] * Rvals[l], gidx, BB);
        long rows = (long)bs * K;
        k_group<<<(int)((rows + 255) / 256), 256, 0, stream>>>(xs[l], ps[l], xs[l + 1],
                                                               gidx, Nl, S, K, Cl, Cp,
                                                               grouped16, BB);
        run_mlp(grouped16, (int)rows, 3 + Cl, Cp, sa[l], Chans[l], 3, actF, nullptr);
        int Cout = Chans[l][2];
        long ptot = (long)bs * Cout;
        k_maxpool<<<(int)((ptot + 255) / 256), 256, 0, stream>>>(actF, bs, K, Cout,
                                                                 ps[l + 1]);
        Nl = S;
    }

    // --- feature propagation ---
    const int fpch4[2] = {256, 256}, fpch3[2] = {256, 256},
              fpch2[2] = {256, 128}, fpch1[3] = {128, 128, 128};
    {   // fp4: 128 <- 32
        int n1 = 128, n2 = 32, c1 = 256, c2 = 512;
        int tot = BB * n1;
        k_fp_interp<<<(tot + 255) / 256, 256, 0, stream>>>(xs[3], xs[4], ps[3], ps[4],
                                                           n1, n2, c1, c2, grouped16, BB);
        run_mlp(grouped16, BB * n1, c1 + c2, c1 + c2, fp4, fpch4, 2, l3, nullptr);
    }
    {   // fp3: 512 <- 128
        int n1 = 512, n2 = 128, c1 = 128, c2 = 256;
        int tot = BB * n1;
        k_fp_interp<<<(tot + 255) / 256, 256, 0, stream>>>(xs[2], xs[3], ps[2], l3,
                                                           n1, n2, c1, c2, grouped16, BB);
        run_mlp(grouped16, BB * n1, c1 + c2, c1 + c2, fp3, fpch3, 2, l2, nullptr);
    }
    {   // fp2: 2048 <- 512
        int n1 = 2048, n2 = 512, c1 = 64, c2 = 256;
        int tot = BB * n1;
        k_fp_interp<<<(tot + 255) / 256, 256, 0, stream>>>(xs[1], xs[2], ps[1], l2,
                                                           n1, n2, c1, c2, grouped16, BB);
        run_mlp(grouped16, BB * n1, c1 + c2, c1 + c2, fp2, fpch2, 2, l1, nullptr);
    }
    {   // fp1: 8192 <- 2048 (no skip features)
        int n1 = 8192, n2 = 2048, c1 = 0, c2 = 128;
        int tot = BB * n1;
        k_fp_interp<<<(tot + 255) / 256, 256, 0, stream>>>(xs[0], xs[1], nullptr, l1,
                                                           n1, n2, c1, c2, grouped16, BB);
        run_mlp(grouped16, BB * n1, 128, 128, fp1, fpch1, 3, l0, l0h);
    }

    // --- head + outputs ---
    {
        const int hch[1] = {19};
        run_mlp(l0h, BB * N0, 128, 128, &head, hch, 1, (float*)d_out, nullptr); // (B,N,19)
        long tot = (long)BB * 128 * N0;
        k_transpose_out<<<(int)((tot + 255) / 256), 256, 0, stream>>>(
            l0, (float*)d_out + (size_t)BB * N0 * 19, BB, N0, 128);             // (B,128,N)
    }
}
